// S3F2Net_24678882083165
// MI455X (gfx1250) — matrix operand order repaired and checked by reading, not compiled, not run on hardware
//
#include <hip/hip_runtime.h>
#include <math.h>

#define EPSV 1e-5f
#define NN 4096
#define HWPIX 16384

typedef __attribute__((ext_vector_type(16))) _Float16 v16h;
typedef __attribute__((ext_vector_type(8)))  _Float16 v8h;
typedef __attribute__((ext_vector_type(8)))  float    v8f;

__device__ __forceinline__ float lrelu_f(float v) { return v >= 0.f ? v : 0.01f * v; }

// ---------------- generic 3x3 SAME conv ----------------
__global__ __launch_bounds__(256) void conv3x3_kernel(
    const float* __restrict__ in, const float* __restrict__ w,
    const float* __restrict__ bias, float* __restrict__ out,
    int B, int Cin, int Cout, int H, int W)
{
  int idx = blockIdx.x * 256 + threadIdx.x;
  int total = B * Cout * H * W;
  if (idx >= total) return;
  int x = idx % W;
  int y = (idx / W) % H;
  int co = (idx / (W * H)) % Cout;
  int b = idx / (W * H * Cout);
  float acc = bias[co];
  for (int ci = 0; ci < Cin; ++ci) {
    const float* ip = in + ((size_t)(b * Cin + ci) * H) * W;
    const float* wp = w + (size_t)(co * Cin + ci) * 9;
    for (int dy = -1; dy <= 1; ++dy) {
      int yy = y + dy;
      if (yy < 0 || yy >= H) continue;
      for (int dx = -1; dx <= 1; ++dx) {
        int xx = x + dx;
        if (xx < 0 || xx >= W) continue;
        acc += ip[yy * W + xx] * wp[(dy + 1) * 3 + (dx + 1)];
      }
    }
  }
  out[idx] = acc;
}

// ---------------- per-channel BN stats (mean, biased var) ----------------
__global__ __launch_bounds__(256) void bnstats_kernel(
    const float* __restrict__ in, float* __restrict__ stats, int B, int C, int HW)
{
  __shared__ float s1[256], s2[256];
  int c = blockIdx.x;
  float sum = 0.f, sq = 0.f;
  int total = B * HW;
  for (int t = threadIdx.x; t < total; t += 256) {
    int b = t / HW, s = t % HW;
    float v = in[(size_t)(b * C + c) * HW + s];
    sum += v; sq += v * v;
  }
  s1[threadIdx.x] = sum; s2[threadIdx.x] = sq;
  __syncthreads();
  for (int st = 128; st > 0; st >>= 1) {
    if (threadIdx.x < st) { s1[threadIdx.x] += s1[threadIdx.x + st]; s2[threadIdx.x] += s2[threadIdx.x + st]; }
    __syncthreads();
  }
  if (threadIdx.x == 0) {
    float m = s1[0] / (float)total;
    stats[c] = m;
    stats[C + c] = s2[0] / (float)total - m * m;
  }
}

// ---------------- BN normalize + (l)relu + 2x2 maxpool ----------------
__global__ __launch_bounds__(256) void bn_act_pool_kernel(
    const float* __restrict__ in, const float* __restrict__ stats,
    float* __restrict__ out, int B, int C, int H, int W, int leaky)
{
  int Ho = H >> 1, Wo = W >> 1;
  int idx = blockIdx.x * 256 + threadIdx.x;
  int total = B * C * Ho * Wo;
  if (idx >= total) return;
  int xo = idx % Wo;
  int yo = (idx / Wo) % Ho;
  int c = (idx / (Wo * Ho)) % C;
  int b = idx / (Wo * Ho * C);
  float m = stats[c];
  float rs = rsqrtf(stats[C + c] + EPSV);
  const float* ip = in + ((size_t)(b * C + c) * H + yo * 2) * W + xo * 2;
  float best = -INFINITY;
  for (int dy = 0; dy < 2; ++dy)
    for (int dx = 0; dx < 2; ++dx) {
      float v = (ip[dy * W + dx] - m) * rs;
      v = leaky ? lrelu_f(v) : fmaxf(v, 0.f);
      best = fmaxf(best, v);
    }
  out[idx] = best;
}

// ---------------- normalize full buffer with stats[0..1] ----------------
__global__ __launch_bounds__(256) void normalize_kernel(
    const float* __restrict__ in, const float* __restrict__ stats,
    float* __restrict__ out, int n)
{
  int idx = blockIdx.x * 256 + threadIdx.x;
  if (idx < n) out[idx] = (in[idx] - stats[0]) * rsqrtf(stats[1] + EPSV);
}

// ---------------- denoise conv output -> [HW,8] flat, lrelu ----------------
__global__ __launch_bounds__(256) void flat_kernel(const float* __restrict__ conv,
                                                   float* __restrict__ flat)
{
  int idx = blockIdx.x * 256 + threadIdx.x;
  if (idx >= HWPIX * 8) return;
  int p = idx / 8, c = idx % 8;
  flat[idx] = lrelu_f(conv[c * HWPIX + p]);
}

// ---------------- feat0 = normQ.T @ flat  (column-per-thread, coalesced) -----
__global__ __launch_bounds__(128) void qtflat_kernel(
    const float* __restrict__ Q, const float* __restrict__ flat,
    float* __restrict__ feat0)
{
  int n = blockIdx.x * 128 + threadIdx.x;
  float acc[8] = {0,0,0,0,0,0,0,0};
  float s = 0.f;
  for (int p = 0; p < HWPIX; ++p) {
    float q = Q[(size_t)p * NN + n];
    if (q != 0.f) {
      s += q;
      const float* f = flat + p * 8;
      for (int c = 0; c < 8; ++c) acc[c] += q * f[c];
    }
  }
  float inv = 1.f / s;
  for (int c = 0; c < 8; ++c) feat0[n * 8 + c] = acc[c] * inv;
}

// ---------------- BatchNorm1d over nodes (single block) ----------------
__global__ __launch_bounds__(256) void bn1d_kernel(
    const float* __restrict__ H, float* __restrict__ Hn, int Nn)
{
  __shared__ float rA[2048], rB[2048];
  __shared__ float sm[8], sr[8];
  float a[8] = {0,0,0,0,0,0,0,0}, q[8] = {0,0,0,0,0,0,0,0};
  for (int n = threadIdx.x; n < Nn; n += 256)
    for (int c = 0; c < 8; ++c) { float v = H[n * 8 + c]; a[c] += v; q[c] += v * v; }
  for (int c = 0; c < 8; ++c) { rA[threadIdx.x * 8 + c] = a[c]; rB[threadIdx.x * 8 + c] = q[c]; }
  __syncthreads();
  for (int st = 128; st > 0; st >>= 1) {
    if (threadIdx.x < st)
      for (int c = 0; c < 8; ++c) {
        rA[threadIdx.x * 8 + c] += rA[(threadIdx.x + st) * 8 + c];
        rB[threadIdx.x * 8 + c] += rB[(threadIdx.x + st) * 8 + c];
      }
    __syncthreads();
  }
  if (threadIdx.x < 8) {
    float m = rA[threadIdx.x] / (float)Nn;
    sm[threadIdx.x] = m;
    sr[threadIdx.x] = rsqrtf(rB[threadIdx.x] / (float)Nn - m * m + EPSV);
  }
  __syncthreads();
  for (int n = threadIdx.x; n < Nn; n += 256)
    for (int c = 0; c < 8; ++c)
      Hn[n * 8 + c] = (H[n * 8 + c] - sm[c]) * sr[c];
}

// ---------------- Hx = Hn @ theta_w.T + theta_b, cast to f16 ----------------
__global__ __launch_bounds__(256) void hx_kernel(
    const float* __restrict__ Hn, const float* __restrict__ tw,
    const float* __restrict__ tb, _Float16* __restrict__ out, int Nn, int K)
{
  int idx = blockIdx.x * 256 + threadIdx.x;
  if (idx >= Nn * K) return;
  int k = idx % K, n = idx / K;
  float acc = tb[k];
  for (int c = 0; c < 8; ++c) acc += Hn[n * 8 + c] * tw[k * 8 + c];
  out[idx] = (_Float16)acc;
}

// ---------------- WMMA: A = clip(sigmoid(Hx @ Hx.T),0.1)*mask + I ----------
// One wave per 16x16 output tile; 8 waves/block -> 64x32 block tile.
// A-frag per ISA 16-bit A 16x32 layout; B-frag per 32x16 layout (lanes = N).
__global__ __launch_bounds__(256) void wmma_sim_kernel(
    const _Float16* __restrict__ Hx, const float* __restrict__ Ain,
    float* __restrict__ Aout, int Nn, int K)
{
  int wave = threadIdx.x >> 5;
  int lane = threadIdx.x & 31;
  int tileRow = blockIdx.x * 4 + (wave & 3);
  int tileCol = blockIdx.y * 2 + (wave >> 2);
  int rowBase = tileRow * 16;
  int colBase = tileCol * 16;
  int r = lane & 15, hi = lane >> 4;
  const _Float16* arow = Hx + (size_t)(rowBase + r) * K;
  const _Float16* brow = Hx + (size_t)(colBase + r) * K;
  v8f acc = {};
  for (int k0 = 0; k0 < K; k0 += 32) {
    v8h a_lo = *(const v8h*)(arow + k0 + hi * 8);
    v8h a_hi = *(const v8h*)(arow + k0 + 16 + hi * 8);
    v8h b_lo = *(const v8h*)(brow + k0 + hi * 8);
    v8h b_hi = *(const v8h*)(brow + k0 + 16 + hi * 8);
    v16h a = __builtin_shufflevector(a_lo, a_hi, 0,1,2,3,4,5,6,7,8,9,10,11,12,13,14,15);
    v16h b = __builtin_shufflevector(b_lo, b_hi, 0,1,2,3,4,5,6,7,8,9,10,11,12,13,14,15);
    acc = __builtin_amdgcn_wmma_f32_16x16x32_f16(false, a, false, b, (short)0, acc, false, false);
  }
  int j = colBase + r;
  #pragma unroll
  for (int v = 0; v < 8; ++v) {
    int i = rowBase + hi * 8 + v;
    float s = acc[v];
    float p = 1.0f / (1.0f + __expf(-s));
    p = fmaxf(p, 0.1f);
    float m = ceilf(Ain[(size_t)i * Nn + j] * 1e-5f);
    Aout[(size_t)i * Nn + j] = p * m + (i == j ? 1.0f : 0.0f);
  }
}

// ---------------- d = rsqrt(A.sum(1)) ----------------
__global__ __launch_bounds__(256) void rowsum_kernel(
    const float* __restrict__ A, float* __restrict__ dvec, int Nn)
{
  __shared__ float red[256];
  int i = blockIdx.x;
  float s = 0.f;
  for (int j = threadIdx.x; j < Nn; j += 256) s += A[(size_t)i * Nn + j];
  red[threadIdx.x] = s;
  __syncthreads();
  for (int st = 128; st > 0; st >>= 1) {
    if (threadIdx.x < st) red[threadIdx.x] += red[threadIdx.x + st];
    __syncthreads();
  }
  if (threadIdx.x == 0) dvec[i] = rsqrtf(red[0]);
}

// ---------------- W[n] = d[n] * (Hn[n] @ out_w.T + out_b) ----------------
__global__ __launch_bounds__(256) void wv_kernel(
    const float* __restrict__ Hn, const float* __restrict__ ow,
    const float* __restrict__ ob, const float* __restrict__ dvec,
    float* __restrict__ W, int Nn)
{
  int idx = blockIdx.x * 256 + threadIdx.x;
  if (idx >= Nn * 8) return;
  int n = idx / 8, c = idx % 8;
  float acc = ob[c];
  for (int k = 0; k < 8; ++k) acc += Hn[n * 8 + k] * ow[c * 8 + k];
  W[idx] = dvec[n] * acc;
}

// ---------------- out[i] = lrelu(d[i] * A[i,:] @ W)  (one row per block) ----
__global__ __launch_bounds__(256) void ahat_matvec_kernel(
    const float* __restrict__ A, const float* __restrict__ W,
    const float* __restrict__ dvec, float* __restrict__ out, int Nn)
{
  __shared__ float red[2048];
  int i = blockIdx.x;
  float acc[8] = {0,0,0,0,0,0,0,0};
  for (int j = threadIdx.x; j < Nn; j += 256) {
    float a = A[(size_t)i * Nn + j];
    const float* w = W + j * 8;
    for (int c = 0; c < 8; ++c) acc[c] += a * w[c];
  }
  for (int c = 0; c < 8; ++c) red[threadIdx.x * 8 + c] = acc[c];
  __syncthreads();
  for (int st = 128; st > 0; st >>= 1) {
    if (threadIdx.x < st)
      for (int c = 0; c < 8; ++c) red[threadIdx.x * 8 + c] += red[(threadIdx.x + st) * 8 + c];
    __syncthreads();
  }
  if (threadIdx.x < 8) out[i * 8 + threadIdx.x] = lrelu_f(dvec[i] * red[threadIdx.x]);
}

// ---------------- generic FC: out[b,o] = feat[b]@w[o] + bias[o] ----------------
__global__ __launch_bounds__(256) void fc_kernel(
    const float* __restrict__ feat, const float* __restrict__ w,
    const float* __restrict__ bias, float* __restrict__ out,
    int Bn, int Cin, int Cout)
{
  int idx = blockIdx.x * 256 + threadIdx.x;
  if (idx >= Bn * Cout) return;
  int o = idx % Cout, b = idx / Cout;
  float acc = bias[o];
  for (int c = 0; c < Cin; ++c) acc += feat[b * Cin + c] * w[o * Cin + c];
  out[idx] = acc;
}

// ---------------- bitonic sort of 4096 scores + th = w2 . sorted + b2 -------
__global__ __launch_bounds__(1024) void sort_th_kernel(
    const float* __restrict__ scores, const float* __restrict__ w2,
    const float* __restrict__ b2, float* __restrict__ thOut)
{
  __shared__ float s[4096];
  __shared__ float red[1024];
  for (int i = threadIdx.x; i < 4096; i += 1024) s[i] = scores[i];
  __syncthreads();
  for (int k = 2; k <= 4096; k <<= 1)
    for (int j = k >> 1; j > 0; j >>= 1) {
      for (int t = threadIdx.x; t < 4096; t += 1024) {
        int partner = t ^ j;
        if (partner > t) {
          bool up = ((t & k) == 0);
          float a = s[t], b = s[partner];
          if ((a > b) == up) { s[t] = b; s[partner] = a; }
        }
      }
      __syncthreads();
    }
  float acc = 0.f;
  for (int i = threadIdx.x; i < 4096; i += 1024) acc += w2[i] * s[i];
  red[threadIdx.x] = acc;
  __syncthreads();
  for (int st = 512; st > 0; st >>= 1) {
    if (threadIdx.x < st) red[threadIdx.x] += red[threadIdx.x + st];
    __syncthreads();
  }
  if (threadIdx.x == 0) thOut[0] = red[0] + b2[0];
}

// recovered[n] = (scores[n] - th) * H1[n]
__global__ __launch_bounds__(256) void recov_kernel(
    const float* __restrict__ scores, const float* __restrict__ th,
    const float* __restrict__ H1, float* __restrict__ rec, int Nn)
{
  int idx = blockIdx.x * 256 + threadIdx.x;
  if (idx >= Nn * 8) return;
  rec[idx] = (scores[idx / 8] - th[0]) * H1[idx];
}

// row L2-normalize
__global__ __launch_bounds__(256) void hn1_kernel(
    const float* __restrict__ H1, float* __restrict__ Hn1, int Nn)
{
  int n = blockIdx.x * 256 + threadIdx.x;
  if (n >= Nn) return;
  float s = 0.f;
  for (int c = 0; c < 8; ++c) { float v = H1[n * 8 + c]; s += v * v; }
  float inv = 1.f / fmaxf(sqrtf(s), 1e-8f);
  for (int c = 0; c < 8; ++c) Hn1[n * 8 + c] = H1[n * 8 + c] * inv;
}

// G[8][8] = Hn1.T @ rec  (rank-8 factor of sim @ recovered)
__global__ __launch_bounds__(64) void gmat_kernel(
    const float* __restrict__ Hn1, const float* __restrict__ rec,
    float* __restrict__ G, int Nn)
{
  int t = threadIdx.x;
  int k = t >> 3, c = t & 7;
  float acc = 0.f;
  for (int n = 0; n < Nn; ++n) acc += Hn1[n * 8 + k] * rec[n * 8 + c];
  G[t] = acc;
}

// updated = Hn1 @ G + rec
__global__ __launch_bounds__(256) void upd_kernel(
    const float* __restrict__ Hn1, const float* __restrict__ G,
    const float* __restrict__ rec, float* __restrict__ out, int Nn)
{
  int idx = blockIdx.x * 256 + threadIdx.x;
  if (idx >= Nn * 8) return;
  int n = idx / 8, c = idx % 8;
  float acc = rec[idx];
  for (int k = 0; k < 8; ++k) acc += Hn1[n * 8 + k] * G[k * 8 + c];
  out[idx] = acc;
}

// monotonic float<->uint mapping for deterministic atomic max
__device__ __forceinline__ unsigned fmap(float f) {
  unsigned u = __float_as_uint(f);
  return (u & 0x80000000u) ? ~u : (u | 0x80000000u);
}

__global__ void init_g8_kernel(unsigned* g8u) {
  if (threadIdx.x < 8 && blockIdx.x == 0) g8u[threadIdx.x] = 0u;
}

// g8 = max over pixels of (Q @ H2)
__global__ __launch_bounds__(256) void qh2max_kernel(
    const float* __restrict__ Q, const float* __restrict__ H2,
    unsigned* __restrict__ g8u, int Nn)
{
  __shared__ float red[2048];
  int p = blockIdx.x;
  float acc[8] = {0,0,0,0,0,0,0,0};
  for (int n = threadIdx.x; n < Nn; n += 256) {
    float q = Q[(size_t)p * Nn + n];
    if (q != 0.f)
      for (int c = 0; c < 8; ++c) acc[c] += q * H2[n * 8 + c];
  }
  for (int c = 0; c < 8; ++c) red[threadIdx.x * 8 + c] = acc[c];
  __syncthreads();
  for (int st = 128; st > 0; st >>= 1) {
    if (threadIdx.x < st)
      for (int c = 0; c < 8; ++c) red[threadIdx.x * 8 + c] += red[(threadIdx.x + st) * 8 + c];
    __syncthreads();
  }
  if (threadIdx.x < 8) atomicMax(&g8u[threadIdx.x], fmap(red[threadIdx.x]));
}

// gvec = softmax(g8 @ lin3_w.T + lin3_b)
__global__ void head_kernel(const unsigned* __restrict__ g8u,
                            const float* __restrict__ w3, const float* __restrict__ b3,
                            float* __restrict__ gvec)
{
  if (threadIdx.x != 0 || blockIdx.x != 0) return;
  float g[8];
  for (int c = 0; c < 8; ++c) {
    unsigned u = g8u[c];
    g[c] = (u & 0x80000000u) ? __uint_as_float(u ^ 0x80000000u) : __uint_as_float(~u);
  }
  float lo[16];
  float mx = -INFINITY;
  for (int o = 0; o < 16; ++o) {
    float acc = b3[o];
    for (int c = 0; c < 8; ++c) acc += g[c] * w3[o * 8 + c];
    lo[o] = acc;
    mx = fmaxf(mx, acc);
  }
  float se = 0.f;
  for (int o = 0; o < 16; ++o) { lo[o] = __expf(lo[o] - mx); se += lo[o]; }
  for (int o = 0; o < 16; ++o) gvec[o] = lo[o] / se;
}

__global__ __launch_bounds__(256) void add_kernel(
    const float* __restrict__ a, const float* __restrict__ b,
    float* __restrict__ o, int n)
{
  int idx = blockIdx.x * 256 + threadIdx.x;
  if (idx < n) o[idx] = a[idx] + b[idx];
}

__global__ __launch_bounds__(256) void final_kernel(
    const float* __restrict__ r1, const float* __restrict__ r2,
    const float* __restrict__ r3, const float* __restrict__ gvec,
    float* __restrict__ out)
{
  int idx = blockIdx.x * 256 + threadIdx.x;
  if (idx >= 3 * 4096) return;
  int which = idx / 4096;
  int rem = idx % 4096;
  int c = rem % 16;
  const float* r = which == 0 ? r1 : (which == 1 ? r2 : r3);
  out[idx] = r[rem] * gvec[c];
}

extern "C" void kernel_launch(void* const* d_in, const int* in_sizes, int n_in,
                              void* d_out, int out_size, void* d_ws, size_t ws_size,
                              hipStream_t stream)
{
  (void)in_sizes; (void)n_in; (void)out_size; (void)ws_size;
  const float* x1 = (const float*)d_in[0];
  const float* x2 = (const float*)d_in[1];
  const float* x3 = (const float*)d_in[2];
  const float* Q  = (const float*)d_in[3];
  const float* Ain = (const float*)d_in[4];
  const float* conv1_w = (const float*)d_in[5];  const float* conv1_b = (const float*)d_in[6];
  const float* conv2_w = (const float*)d_in[7];  const float* conv2_b = (const float*)d_in[8];
  const float* conv3_w = (const float*)d_in[9];  const float* conv3_b = (const float*)d_in[10];
  const float* conv4_w = (const float*)d_in[11]; const float* conv4_b = (const float*)d_in[12];
  const float* conv5_w = (const float*)d_in[13]; const float* conv5_b = (const float*)d_in[14];
  const float* conv6_w = (const float*)d_in[15]; const float* conv6_b = (const float*)d_in[16];
  const float* conv7_w = (const float*)d_in[17]; const float* conv7_b = (const float*)d_in[18];
  const float* den_w = (const float*)d_in[19];   const float* den_b = (const float*)d_in[20];
  const float* out1_w = (const float*)d_in[21];  const float* out1_b = (const float*)d_in[22];
  const float* out2_w = (const float*)d_in[23];  const float* out2_b = (const float*)d_in[24];
  const float* out3_w = (const float*)d_in[25];  const float* out3_b = (const float*)d_in[26];
  const float* g1_tw = (const float*)d_in[27];   const float* g1_tb = (const float*)d_in[28];
  const float* g1_ow = (const float*)d_in[29];   const float* g1_ob = (const float*)d_in[30];
  const float* g2_tw = (const float*)d_in[31];   const float* g2_tb = (const float*)d_in[32];
  const float* g2_ow = (const float*)d_in[33];   const float* g2_ob = (const float*)d_in[34];
  const float* lin1_w = (const float*)d_in[35];  const float* lin1_b = (const float*)d_in[36];
  const float* lin2_w = (const float*)d_in[37];  const float* lin2_b = (const float*)d_in[38];
  const float* lin3_w = (const float*)d_in[39];  const float* lin3_b = (const float*)d_in[40];

  char* base = (char*)d_ws;
  size_t off = 0;
  auto alloc = [&](size_t bytes) -> void* {
    void* p = base + off;
    off += (bytes + 255) & ~(size_t)255;
    return p;
  };
  float* wsA = (float*)alloc((size_t)NN * NN * 4);          // 64 MB, L2-resident
  _Float16* hx16 = (_Float16*)alloc((size_t)NN * 256 * 2);  // 2 MB
  float* dvec = (float*)alloc(NN * 4);
  float* stats = (float*)alloc(1024);
  float* feat0 = (float*)alloc(NN * 8 * 4);
  float* Hn = (float*)alloc(NN * 8 * 4);
  float* Wb = (float*)alloc(NN * 8 * 4);
  float* H1 = (float*)alloc(NN * 8 * 4);
  float* Hn1 = (float*)alloc(NN * 8 * 4);
  float* rec = (float*)alloc(NN * 8 * 4);
  float* updv = (float*)alloc(NN * 8 * 4);
  float* H2 = (float*)alloc(NN * 8 * 4);
  float* scoresv = (float*)alloc(NN * 4);
  float* thv = (float*)alloc(256);
  float* Gm = (float*)alloc(64 * 4);
  unsigned* g8u = (unsigned*)alloc(256);
  float* gvec = (float*)alloc(64 * 4);
  float* t0 = (float*)alloc((size_t)256 * 32 * 64 * 4);  // 2 MB conv scratch
  float* t1 = (float*)alloc((size_t)256 * 64 * 16 * 4);
  float* t2 = (float*)alloc((size_t)256 * 128 * 4 * 4);
  float* c1 = (float*)alloc((size_t)256 * 32 * 16 * 4);
  float* c4 = (float*)alloc((size_t)256 * 32 * 16 * 4);
  float* c2 = (float*)alloc((size_t)256 * 64 * 4 * 4);
  float* c5 = (float*)alloc((size_t)256 * 64 * 4 * 4);
  float* xh = (float*)alloc(256 * 128 * 4);
  float* xl = (float*)alloc(256 * 128 * 4);
  float* x1f = (float*)alloc(256 * 128 * 4);
  float* x2f = (float*)alloc(256 * 128 * 4);
  float* fus = (float*)alloc(256 * 128 * 4);
  float* xdn = (float*)alloc(HWPIX * 4);
  float* flat = (float*)alloc(HWPIX * 8 * 4);
  float* r1 = (float*)alloc(256 * 16 * 4);
  float* r2 = (float*)alloc(256 * 16 * 4);
  float* r3 = (float*)alloc(256 * 16 * 4);

  // ------- CNN branch 1 -------
  conv3x3_kernel<<<2048, 256, 0, stream>>>(x1, conv1_w, conv1_b, t0, 256, 64, 32, 8, 8);
  bnstats_kernel<<<32, 256, 0, stream>>>(t0, stats, 256, 32, 64);
  bn_act_pool_kernel<<<512, 256, 0, stream>>>(t0, stats, c1, 256, 32, 8, 8, 0);
  conv3x3_kernel<<<1024, 256, 0, stream>>>(c1, conv2_w, conv2_b, t1, 256, 32, 64, 4, 4);
  bnstats_kernel<<<64, 256, 0, stream>>>(t1, stats, 256, 64, 16);
  bn_act_pool_kernel<<<256, 256, 0, stream>>>(t1, stats, c2, 256, 64, 4, 4, 0);
  conv3x3_kernel<<<512, 256, 0, stream>>>(c2, conv3_w, conv3_b, t2, 256, 64, 128, 2, 2);
  bnstats_kernel<<<128, 256, 0, stream>>>(t2, stats, 256, 128, 4);
  bn_act_pool_kernel<<<128, 256, 0, stream>>>(t2, stats, xh, 256, 128, 2, 2, 0);
  // ------- CNN branch 2 -------
  conv3x3_kernel<<<2048, 256, 0, stream>>>(x2, conv4_w, conv4_b, t0, 256, 1, 32, 8, 8);
  bnstats_kernel<<<32, 256, 0, stream>>>(t0, stats, 256, 32, 64);
  bn_act_pool_kernel<<<512, 256, 0, stream>>>(t0, stats, c4, 256, 32, 8, 8, 0);
  conv3x3_kernel<<<1024, 256, 0, stream>>>(c4, conv5_w, conv5_b, t1, 256, 32, 64, 4, 4);
  bnstats_kernel<<<64, 256, 0, stream>>>(t1, stats, 256, 64, 16);
  bn_act_pool_kernel<<<256, 256, 0, stream>>>(t1, stats, c5, 256, 64, 4, 4, 1);  // lrelu
  conv3x3_kernel<<<512, 256, 0, stream>>>(c5, conv6_w, conv6_b, t2, 256, 64, 128, 2, 2);
  bnstats_kernel<<<128, 256, 0, stream>>>(t2, stats, 256, 128, 4);
  bn_act_pool_kernel<<<128, 256, 0, stream>>>(t2, stats, xl, 256, 128, 2, 2, 0);
  // ------- shared conv7 on both paths -------
  conv3x3_kernel<<<512, 256, 0, stream>>>(c2, conv7_w, conv7_b, t2, 256, 64, 128, 2, 2);
  bnstats_kernel<<<128, 256, 0, stream>>>(t2, stats, 256, 128, 4);
  bn_act_pool_kernel<<<128, 256, 0, stream>>>(t2, stats, x1f, 256, 128, 2, 2, 0);
  conv3x3_kernel<<<512, 256, 0, stream>>>(c5, conv7_w, conv7_b, t2, 256, 64, 128, 2, 2);
  bnstats_kernel<<<128, 256, 0, stream>>>(t2, stats, 256, 128, 4);
  bn_act_pool_kernel<<<128, 256, 0, stream>>>(t2, stats, x2f, 256, 128, 2, 2, 0);
  add_kernel<<<128, 256, 0, stream>>>(x1f, x2f, fus, 256 * 128);
  fc_kernel<<<16, 256, 0, stream>>>(xh, out1_w, out1_b, r1, 256, 128, 16);
  fc_kernel<<<16, 256, 0, stream>>>(xl, out2_w, out2_b, r2, 256, 128, 16);
  fc_kernel<<<16, 256, 0, stream>>>(fus, out3_w, out3_b, r3, 256, 128, 16);

  // ------- denoise -> pixel features -------
  bnstats_kernel<<<1, 256, 0, stream>>>(x3, stats, 1, 1, HWPIX);
  normalize_kernel<<<64, 256, 0, stream>>>(x3, stats, xdn, HWPIX);
  conv3x3_kernel<<<512, 256, 0, stream>>>(xdn, den_w, den_b, t0, 1, 1, 8, 128, 128);
  flat_kernel<<<512, 256, 0, stream>>>(t0, flat);
  qtflat_kernel<<<32, 128, 0, stream>>>(Q, flat, feat0);

  // ------- GCN helper sequence -------
  auto run_gcn = [&](const float* Hin, const float* tw, const float* tb,
                     const float* ow, const float* ob, float* Hout) {
    bn1d_kernel<<<1, 256, 0, stream>>>(Hin, Hn, NN);
    hx_kernel<<<4096, 256, 0, stream>>>(Hn, tw, tb, hx16, NN, 256);
    dim3 g(64, 128);
    wmma_sim_kernel<<<g, 256, 0, stream>>>(hx16, Ain, wsA, NN, 256);
    rowsum_kernel<<<NN, 256, 0, stream>>>(wsA, dvec, NN);
    wv_kernel<<<128, 256, 0, stream>>>(Hn, ow, ob, dvec, Wb, NN);
    ahat_matvec_kernel<<<NN, 256, 0, stream>>>(wsA, Wb, dvec, Hout, NN);
  };

  run_gcn(feat0, g1_tw, g1_tb, g1_ow, g1_ob, H1);

  // scores / threshold / recovered / rank-8 sim update
  fc_kernel<<<16, 256, 0, stream>>>(H1, lin1_w, lin1_b, scoresv, NN, 8, 1);
  sort_th_kernel<<<1, 1024, 0, stream>>>(scoresv, lin2_w, lin2_b, thv);
  recov_kernel<<<128, 256, 0, stream>>>(scoresv, thv, H1, rec, NN);
  hn1_kernel<<<16, 256, 0, stream>>>(H1, Hn1, NN);
  gmat_kernel<<<1, 64, 0, stream>>>(Hn1, rec, Gm, NN);
  upd_kernel<<<128, 256, 0, stream>>>(Hn1, Gm, rec, updv, NN);

  run_gcn(updv, g2_tw, g2_tb, g2_ow, g2_ob, H2);

  // ------- pooled head + softmax gate -------
  init_g8_kernel<<<1, 32, 0, stream>>>(g8u);
  qh2max_kernel<<<HWPIX, 256, 0, stream>>>(Q, H2, g8u, NN);
  head_kernel<<<1, 32, 0, stream>>>(g8u, lin3_w, lin3_b, gvec);
  final_kernel<<<48, 256, 0, stream>>>(r1, r2, r3, gvec, (float*)d_out);
}
